// TransformerBlock_16999480557970
// MI455X (gfx1250) — compile-verified, matrix operand order checked by reading, not run on hardware
//
#include <hip/hip_runtime.h>
#include <hip/hip_bf16.h>

// ---------------------------------------------------------------------------
// CDNA5 (gfx1250) WMMA implementation of a PointTransformerConv block.
// All 64x64 GEMMs run on v_wmma_f32_16x16x32_f16 (f16 inputs, f32 accumulate).
// Weights are pre-swizzled into B-fragment order (contiguous 32B per lane);
// intermediates live in LDS as f16 tiles in A-fragment-friendly K-permuted
// layout so every fragment load is a single contiguous v16h (2x ds_load_b128).
// GEMM kernels use capped grids + uniform-trip tile loops so the LDS weight
// staging is amortized over many tiles per block.
// Scatter softmax / segment sums stay in fp32 VALU + global float atomics.
// ---------------------------------------------------------------------------

typedef __attribute__((ext_vector_type(16))) _Float16 v16h;
typedef __attribute__((ext_vector_type(8)))  float    v8f;
typedef __attribute__((ext_vector_type(4)))  _Float16 v4h;

#define WMMA_F16(a, b, c) \
  __builtin_amdgcn_wmma_f32_16x16x32_f16(false, (a), false, (b), (short)0, (c), false, false)

#define WAVES 4
#define BLOCK (WAVES * 32)
#define TSH   80          // f16 tile row stride in halves (160B = 32B multiple)
#define MATSZ 4096        // halves per swizzled 64x64 matrix (2 kt * 4 ct * 32 lanes * 16)

#define NODE_GRID_CAP 512
#define EDGE_GRID_CAP 1024

// K permutation for the f16 LDS tiles: element K=c of a row is stored at
// perm64(c) so that lane (row, hi)'s A fragment {K: kmap(j,hi)} is contiguous.
__device__ __forceinline__ int perm64(int c) {
  const int kt = c >> 5, kk = c & 31;
  int p;
  if (kk & 16) p = (kk & 8) ? kk : kk - 8;
  else         p = (kk & 8) ? kk + 8 : kk;
  return kt * 32 + p;
}

// A fragment: one contiguous 32B load per lane from the permuted f16 tile.
__device__ __forceinline__ v16h afrag16(const _Float16* tile, int lane, int kt) {
  return *(const v16h*)(tile + (lane & 15) * TSH + kt * 32 + (lane >> 4) * 16);
}

// D = A[16x64] * W[64x64] with pre-swizzled weights: B fragment = contiguous
// 32B per lane, 8 wmma ops total (4 col tiles x 2 K tiles).
__device__ __forceinline__ void gemm16x64(const v16h a0, const v16h a1,
                                          const _Float16* wswz, int lane, v8f acc[4]) {
#pragma unroll
  for (int ct = 0; ct < 4; ++ct) {
    const v16h b0 = *(const v16h*)(wswz + ((0 * 4 + ct) * 32 + lane) * 16);
    const v16h b1 = *(const v16h*)(wswz + ((1 * 4 + ct) * 32 + lane) * 16);
    v8f c = {};
    c = WMMA_F16(a0, b0, c);
    c = WMMA_F16(a1, b1, c);
    acc[ct] = c;
  }
}

// Write D fragments (lane holds col, vgpr r holds row r + 8*hi) into the
// permuted f16 LDS tile, with optional bias + relu.
__device__ __forceinline__ void dstore_lds16(_Float16* tile, int lane, const v8f acc[4],
                                             const float* bias, bool relu) {
  const int rbase = (lane >> 4) * 8, coll = lane & 15;
#pragma unroll
  for (int ct = 0; ct < 4; ++ct) {
    const int col = coll + ct * 16;
    const int pc  = perm64(col);
    const float bb = bias ? bias[col] : 0.0f;
#pragma unroll
    for (int r = 0; r < 8; ++r) {
      float vv = acc[ct][r] + bb;
      if (relu) vv = fmaxf(vv, 0.0f);
      tile[(rbase + r) * TSH + pc] = (_Float16)vv;
    }
  }
}

// Write D fragments to a [rows,64] row-major f32 global array (row-clamped).
__device__ __forceinline__ void dstore_global(float* g, int rowbase, int rowlimit,
                                              const v8f acc[4], int lane,
                                              const float* bias, bool relu) {
  const int rbase = (lane >> 4) * 8, coll = lane & 15;
#pragma unroll
  for (int ct = 0; ct < 4; ++ct) {
    const int col = coll + ct * 16;
    const float bb = bias ? bias[col] : 0.0f;
#pragma unroll
    for (int r = 0; r < 8; ++r) {
      int row = rowbase + rbase + r;
      if (row >= rowlimit) row = rowlimit - 1;
      float vv = acc[ct][r] + bb;
      if (relu) vv = fmaxf(vv, 0.0f);
      g[(size_t)row * 64 + col] = vv;
    }
  }
}

// Stage a [16,64] f32 global tile into the permuted f16 LDS tile.
// 4-channel packets: float4 global load + 8B LDS store (perm keeps runs of 4).
__device__ __forceinline__ void stage_tile(const float* __restrict__ g, int rowbase,
                                           int rowlimit, _Float16* tile, int lane) {
  for (int i = lane; i < 16 * 16; i += 32) {
    const int r = i >> 4, c = (i & 15) * 4;
    int row = rowbase + r; if (row >= rowlimit) row = rowlimit - 1;
    const float4 vv = *(const float4*)(g + (size_t)row * 64 + c);
    v4h p; p[0] = (_Float16)vv.x; p[1] = (_Float16)vv.y;
           p[2] = (_Float16)vv.z; p[3] = (_Float16)vv.w;
    *(v4h*)(tile + r * TSH + perm64(c)) = p;
  }
}

__device__ __forceinline__ void atomicMaxF32(float* addr, float val) {
  __hip_atomic_fetch_max(addr, val, __ATOMIC_RELAXED, __HIP_MEMORY_SCOPE_AGENT);
}

// ---------------------------------------------------------------------------
// 0) fp32 -> f16 weight conversion + swizzle into B-fragment order.
//    layout: mat m, ktile kt, coltile ct, lane l, j ->
//    w16s[((m*2+kt)*4+ct)*512 + l*16 + j] = W_m[kt*32+kmap(j,l>=16)][ct*16+(l&15)]
// ---------------------------------------------------------------------------
__global__ void prep_weights_kernel(const float* __restrict__ w0, const float* __restrict__ w1,
                                    const float* __restrict__ w2, const float* __restrict__ w3,
                                    const float* __restrict__ w4, const float* __restrict__ w5,
                                    const float* __restrict__ w6, const float* __restrict__ w7,
                                    _Float16* __restrict__ w16s) {
  const int i = blockIdx.x * blockDim.x + threadIdx.x;
  if (i >= 8 * MATSZ) return;
  const int m    = i >> 12;
  const int r    = i & (MATSZ - 1);
  const int j    = r & 15;
  const int lane = (r >> 4) & 31;
  const int ct   = (r >> 9) & 3;
  const int kt   = (r >> 11) & 1;
  const int hi   = lane >> 4;
  const int k    = kt * 32 + ((j < 8) ? j : j + 8) + (hi ? 8 : 0);
  const int col  = ct * 16 + (lane & 15);
  const float* srcs[8] = {w0, w1, w2, w3, w4, w5, w6, w7};
  w16s[i] = (_Float16)srcs[m][k * 64 + col];
}

// ---------------------------------------------------------------------------
// 1) init segment-max / segment-sum / output accumulator.
// ---------------------------------------------------------------------------
__global__ void init_kernel(float* __restrict__ m, float* __restrict__ s,
                            float* __restrict__ accum, int nc) {
  int i = blockIdx.x * blockDim.x + threadIdx.x;
  if (i >= nc) return;
  m[i] = -3.0e38f;
  s[i] = 0.0f;
  accum[i] = 0.0f;
}

// ---------------------------------------------------------------------------
// 2) node projections: h = relu(x W_in + b); v = h W_lin; a_src = h W_src;
//    a_dst = h W_dst.  4 WMMA GEMMs per 16-node tile; tile loop per wave.
// ---------------------------------------------------------------------------
__global__ __launch_bounds__(BLOCK) void node_proj_kernel(
    const float* __restrict__ x, const _Float16* __restrict__ w16s,
    const float* __restrict__ b_in,
    float* __restrict__ v_out, float* __restrict__ asrc_out, float* __restrict__ adst_out,
    int N) {
  __shared__ __align__(32) _Float16 sW[4 * MATSZ];          // W_in, W_lin, W_src, W_dst
  __shared__ __align__(32) _Float16 sTile[WAVES][16 * TSH];

  const int tid = threadIdx.x, lane = tid & 31, wave = tid >> 5;
  for (int i = tid; i < 4 * MATSZ / 4; i += BLOCK)          // 8B-vector weight copy
    ((uint2*)sW)[i] = ((const uint2*)w16s)[i];
  __syncthreads();

  const int numTiles     = (N + 15) >> 4;
  const int tilesPerGrid = gridDim.x * WAVES;
  const int numIter      = (numTiles + tilesPerGrid - 1) / tilesPerGrid;  // uniform trips
  _Float16* myTile = sTile[wave];

  for (int it = 0; it < numIter; ++it) {
    int tile = it * tilesPerGrid + blockIdx.x * WAVES + wave;
    if (tile >= numTiles) tile = numTiles - 1;              // keep EXEC all-ones for WMMA
    const int rowbase = tile * 16;

    stage_tile(x, rowbase, N, myTile, lane);
    __syncthreads();

    v16h a0 = afrag16(myTile, lane, 0);
    v16h a1 = afrag16(myTile, lane, 1);
    v8f acc[4];
    gemm16x64(a0, a1, sW, lane, acc);                       // x @ W_in
    __syncthreads();
    dstore_lds16(myTile, lane, acc, b_in, true);            // h = relu(. + b_in)
    __syncthreads();

    v16h h0 = afrag16(myTile, lane, 0);
    v16h h1 = afrag16(myTile, lane, 1);
    gemm16x64(h0, h1, sW + 1 * MATSZ, lane, acc);
    dstore_global(v_out, rowbase, N, acc, lane, nullptr, false);
    gemm16x64(h0, h1, sW + 2 * MATSZ, lane, acc);
    dstore_global(asrc_out, rowbase, N, acc, lane, nullptr, false);
    gemm16x64(h0, h1, sW + 3 * MATSZ, lane, acc);
    dstore_global(adst_out, rowbase, N, acc, lane, nullptr, false);
    __syncthreads();
  }
}

// ---------------------------------------------------------------------------
// 3) edge MLPs: delta = relu(rel Wp1 + bp1) Wp2 + bp2
//    logit = relu((a_dst[d]-a_src[s]+delta) Wa1 + ba1) Wa2 + ba2
//    plus atomic per-(dst,channel) max for the softmax.  Tile loop per wave.
// ---------------------------------------------------------------------------
__global__ __launch_bounds__(BLOCK) void edge_mlp_kernel(
    const float* __restrict__ pos, const int* __restrict__ edge_index,
    const float* __restrict__ asrc, const float* __restrict__ adst,
    const _Float16* __restrict__ w16s,
    const float* __restrict__ Wp1, const float* __restrict__ bp1,
    const float* __restrict__ bp2, const float* __restrict__ ba1,
    const float* __restrict__ ba2,
    float* __restrict__ delta_ws, float* __restrict__ logit_ws, float* __restrict__ m_ws,
    int E) {
  __shared__ __align__(32) _Float16 sW[3 * MATSZ];          // Wp2, Wa1, Wa2 (swizzled)
  __shared__ float    sWp1[192];
  __shared__ __align__(32) _Float16 sTile[WAVES][16 * TSH];
  __shared__ int      sSrc[WAVES][16], sDst[WAVES][16];
  __shared__ float    sRel[WAVES][48];

  const int tid = threadIdx.x, lane = tid & 31, wave = tid >> 5;
  for (int i = tid; i < 3 * MATSZ / 4; i += BLOCK)
    ((uint2*)sW)[i] = ((const uint2*)(w16s + 4 * MATSZ))[i];
  for (int i = tid; i < 192; i += BLOCK) sWp1[i] = Wp1[i];
  __syncthreads();

  const int numTiles     = (E + 15) >> 4;
  const int tilesPerGrid = gridDim.x * WAVES;
  const int numIter      = (numTiles + tilesPerGrid - 1) / tilesPerGrid;  // uniform trips
  _Float16* myTile = sTile[wave];
  const int row = lane & 15, hi = lane >> 4;

  for (int it = 0; it < numIter; ++it) {
    int tile = it * tilesPerGrid + blockIdx.x * WAVES + wave;
    if (tile >= numTiles) tile = numTiles - 1;              // keep EXEC all-ones for WMMA
    const int eb = tile * 16;

    if (lane < 16) {
      int e = eb + lane; if (e >= E) e = E - 1;
      const int s = edge_index[e];
      const int d = edge_index[E + e];
      sSrc[wave][lane] = s;
      sDst[wave][lane] = d;
      sRel[wave][lane * 3 + 0] = pos[(size_t)d * 3 + 0] - pos[(size_t)s * 3 + 0];
      sRel[wave][lane * 3 + 1] = pos[(size_t)d * 3 + 1] - pos[(size_t)s * 3 + 1];
      sRel[wave][lane * 3 + 2] = pos[(size_t)d * 3 + 2] - pos[(size_t)s * 3 + 2];
    }
    __syncthreads();

    // t = relu(rel @ Wp1 + bp1) built directly in A-fragment layout (K=3: VALU)
    const float r0 = sRel[wave][row * 3 + 0];
    const float r1 = sRel[wave][row * 3 + 1];
    const float r2 = sRel[wave][row * 3 + 2];
    v16h t0, t1;
#pragma unroll
    for (int j = 0; j < 16; ++j) {
      int k = ((j < 8) ? j : j + 8) + (hi ? 8 : 0);
      float tv = fmaf(r0, sWp1[k], fmaf(r1, sWp1[64 + k], fmaf(r2, sWp1[128 + k], bp1[k])));
      t0[j] = (_Float16)fmaxf(tv, 0.0f);
      k += 32;
      tv = fmaf(r0, sWp1[k], fmaf(r1, sWp1[64 + k], fmaf(r2, sWp1[128 + k], bp1[k])));
      t1[j] = (_Float16)fmaxf(tv, 0.0f);
    }

    // delta = t @ Wp2 + bp2  -> global (f32) + permuted LDS tile (f16)
    v8f acc[4];
    gemm16x64(t0, t1, sW, lane, acc);
    {
      const int rbase = (lane >> 4) * 8, coll = lane & 15;
#pragma unroll
      for (int ct = 0; ct < 4; ++ct) {
        const int col = coll + ct * 16;
        const int pc  = perm64(col);
        const float bb = bp2[col];
#pragma unroll
        for (int r = 0; r < 8; ++r) {
          int e = eb + rbase + r; if (e >= E) e = E - 1;
          const float vv = acc[ct][r] + bb;
          delta_ws[(size_t)e * 64 + col] = vv;
          myTile[(rbase + r) * TSH + pc] = (_Float16)vv;
        }
      }
    }
    __syncthreads();

    // comb = a_dst[dst] - a_src[src] + delta  (4-channel packets, in-place)
    for (int i = lane; i < 16 * 16; i += 32) {
      const int e = i >> 4, c = (i & 15) * 4;
      const int s = sSrc[wave][e], d = sDst[wave][e];
      const float4 ad = *(const float4*)(adst + (size_t)d * 64 + c);
      const float4 as = *(const float4*)(asrc + (size_t)s * 64 + c);
      v4h* p = (v4h*)(myTile + e * TSH + perm64(c));
      v4h dv = *p, o;
      o[0] = (_Float16)((float)dv[0] + ad.x - as.x);
      o[1] = (_Float16)((float)dv[1] + ad.y - as.y);
      o[2] = (_Float16)((float)dv[2] + ad.z - as.z);
      o[3] = (_Float16)((float)dv[3] + ad.w - as.w);
      *p = o;
    }
    __syncthreads();

    v16h c0 = afrag16(myTile, lane, 0);
    v16h c1 = afrag16(myTile, lane, 1);
    gemm16x64(c0, c1, sW + 1 * MATSZ, lane, acc);           // comb @ Wa1
    __syncthreads();
    dstore_lds16(myTile, lane, acc, ba1, true);             // u = relu(. + ba1)
    __syncthreads();

    v16h u0 = afrag16(myTile, lane, 0);
    v16h u1 = afrag16(myTile, lane, 1);
    gemm16x64(u0, u1, sW + 2 * MATSZ, lane, acc);           // logits = u @ Wa2 + ba2
    {
      const int rbase = (lane >> 4) * 8, coll = lane & 15;
#pragma unroll
      for (int ct = 0; ct < 4; ++ct) {
        const int col = coll + ct * 16;
        const float bb = ba2[col];
#pragma unroll
        for (int r = 0; r < 8; ++r) {
          int e = eb + rbase + r; if (e >= E) e = E - 1;
          const float lv = acc[ct][r] + bb;
          logit_ws[(size_t)e * 64 + col] = lv;
          const int d = sDst[wave][rbase + r];
          atomicMaxF32(&m_ws[(size_t)d * 64 + col], lv);    // segment max
        }
      }
    }
    __syncthreads();
  }
}

// ---------------------------------------------------------------------------
// 4) segment softmax denominator: p = exp(logit - m[d,c]); s[d,c] += p;
//    p is written back over logit_ws so the aggregation pass needs neither
//    the m gather nor a second exp.
// ---------------------------------------------------------------------------
__global__ void softmax_sum_kernel(float* __restrict__ logit, const float* __restrict__ m,
                                   const int* __restrict__ edge_index,
                                   float* __restrict__ ssum, long long EC, int E) {
  long long i = (long long)blockIdx.x * blockDim.x + threadIdx.x;
  if (i >= EC) return;
  const int e = (int)(i >> 6), c = (int)(i & 63);
  const int d = edge_index[E + e];
  const size_t dc = (size_t)d * 64 + c;
  const float p = __expf(logit[i] - m[dc]);
  logit[i] = p;
  atomicAdd(&ssum[dc], p);
}

// ---------------------------------------------------------------------------
// 5) aggregation: accum[d,c] += (p / s[d,c]) * (v[s,c] + delta[e,c])
// ---------------------------------------------------------------------------
__global__ void aggregate_kernel(const float* __restrict__ pexp, const float* __restrict__ delta,
                                 const float* __restrict__ ssum,
                                 const float* __restrict__ v, const int* __restrict__ edge_index,
                                 float* __restrict__ accum, long long EC, int E) {
  long long i = (long long)blockIdx.x * blockDim.x + threadIdx.x;
  if (i >= EC) return;
  const int e = (int)(i >> 6), c = (int)(i & 63);
  const int s = edge_index[e];
  const int d = edge_index[E + e];
  const size_t dc = (size_t)d * 64 + c;
  const float p = pexp[i] / (ssum[dc] + 1e-16f);
  atomicAdd(&accum[dc], p * (v[(size_t)s * 64 + c] + delta[i]));
}

// ---------------------------------------------------------------------------
// 6) output projection: out = relu(accum @ W_out + b_out); tile loop per wave.
// ---------------------------------------------------------------------------
__global__ __launch_bounds__(BLOCK) void out_proj_kernel(
    const float* __restrict__ accum, const _Float16* __restrict__ w16s,
    const float* __restrict__ b_out, float* __restrict__ out, int N) {
  __shared__ __align__(32) _Float16 sW[MATSZ];              // W_out (swizzled)
  __shared__ __align__(32) _Float16 sTile[WAVES][16 * TSH];

  const int tid = threadIdx.x, lane = tid & 31, wave = tid >> 5;
  for (int i = tid; i < MATSZ / 4; i += BLOCK)
    ((uint2*)sW)[i] = ((const uint2*)(w16s + 7 * MATSZ))[i];
  __syncthreads();

  const int numTiles     = (N + 15) >> 4;
  const int tilesPerGrid = gridDim.x * WAVES;
  const int numIter      = (numTiles + tilesPerGrid - 1) / tilesPerGrid;
  _Float16* myTile = sTile[wave];

  for (int it = 0; it < numIter; ++it) {
    int tile = it * tilesPerGrid + blockIdx.x * WAVES + wave;
    if (tile >= numTiles) tile = numTiles - 1;
    const int rowbase = tile * 16;

    stage_tile(accum, rowbase, N, myTile, lane);
    __syncthreads();

    v16h a0 = afrag16(myTile, lane, 0);
    v16h a1 = afrag16(myTile, lane, 1);
    v8f acc[4];
    gemm16x64(a0, a1, sW, lane, acc);
    dstore_global(out, rowbase, N, acc, lane, b_out, true);
    __syncthreads();
  }
}

// ---------------------------------------------------------------------------
// launcher
// ---------------------------------------------------------------------------
extern "C" void kernel_launch(void* const* d_in, const int* in_sizes, int n_in,
                              void* d_out, int out_size, void* d_ws, size_t ws_size,
                              hipStream_t stream) {
  // inputs (setup_inputs order)
  const float* x          = (const float*)d_in[0];
  const float* pos        = (const float*)d_in[1];
  const int*   edge_index = (const int*)  d_in[2];
  const float* W_in  = (const float*)d_in[3];   const float* b_in  = (const float*)d_in[4];
  const float* W_lin = (const float*)d_in[5];
  const float* W_src = (const float*)d_in[6];
  const float* W_dst = (const float*)d_in[7];
  const float* Wp1   = (const float*)d_in[8];   const float* bp1   = (const float*)d_in[9];
  const float* Wp2   = (const float*)d_in[10];  const float* bp2   = (const float*)d_in[11];
  const float* Wa1   = (const float*)d_in[12];  const float* ba1   = (const float*)d_in[13];
  const float* Wa2   = (const float*)d_in[14];  const float* ba2   = (const float*)d_in[15];
  const float* W_out = (const float*)d_in[16];  const float* b_out = (const float*)d_in[17];

  const int N = in_sizes[0] / 64;
  const int E = in_sizes[2] / 2;
  const long long EC = (long long)E * 64;

  // workspace layout
  char* ws = (char*)d_ws;
  _Float16* w16s = (_Float16*)ws;
  size_t off = (size_t)8 * MATSZ * sizeof(_Float16);
  float* v_ws     = (float*)(ws + off); off += (size_t)N * 64 * sizeof(float);
  float* asrc_ws  = (float*)(ws + off); off += (size_t)N * 64 * sizeof(float);
  float* adst_ws  = (float*)(ws + off); off += (size_t)N * 64 * sizeof(float);
  float* m_ws     = (float*)(ws + off); off += (size_t)N * 64 * sizeof(float);
  float* s_ws     = (float*)(ws + off); off += (size_t)N * 64 * sizeof(float);
  float* accum_ws = (float*)(ws + off); off += (size_t)N * 64 * sizeof(float);
  float* delta_ws = (float*)(ws + off); off += (size_t)E * 64 * sizeof(float);
  float* logit_ws = (float*)(ws + off); off += (size_t)E * 64 * sizeof(float);
  (void)ws_size; (void)n_in; (void)out_size;

  // 0) weight conversion + swizzle
  prep_weights_kernel<<<(8 * MATSZ + 255) / 256, 256, 0, stream>>>(
      W_in, W_lin, W_src, W_dst, Wp2, Wa1, Wa2, W_out, w16s);

  // 1) init softmax stats + output accumulator
  const int NC = N * 64;
  init_kernel<<<(NC + 255) / 256, 256, 0, stream>>>(m_ws, s_ws, accum_ws, NC);

  // 2) node projections (capped grid + per-wave tile loop)
  const int nodeTiles  = (N + 15) / 16;
  int nodeBlocks = (nodeTiles + WAVES - 1) / WAVES;
  if (nodeBlocks > NODE_GRID_CAP) nodeBlocks = NODE_GRID_CAP;
  node_proj_kernel<<<nodeBlocks, BLOCK, 0, stream>>>(
      x, w16s, b_in, v_ws, asrc_ws, adst_ws, N);

  // 3) edge MLPs + segment max (capped grid + per-wave tile loop)
  const int edgeTiles = (E + 15) / 16;
  int edgeBlocks = (edgeTiles + WAVES - 1) / WAVES;
  if (edgeBlocks > EDGE_GRID_CAP) edgeBlocks = EDGE_GRID_CAP;
  edge_mlp_kernel<<<edgeBlocks, BLOCK, 0, stream>>>(
      pos, edge_index, asrc_ws, adst_ws, w16s, Wp1, bp1, bp2, ba1, ba2,
      delta_ws, logit_ws, m_ws, E);

  // 4) softmax denominator (writes exp back over logit_ws)
  softmax_sum_kernel<<<(unsigned)((EC + 255) / 256), 256, 0, stream>>>(
      logit_ws, m_ws, edge_index, s_ws, EC, E);

  // 5) weighted aggregation
  aggregate_kernel<<<(unsigned)((EC + 255) / 256), 256, 0, stream>>>(
      logit_ws, delta_ws, s_ws, v_ws, edge_index, accum_ws, EC, E);

  // 6) output projection
  out_proj_kernel<<<nodeBlocks, BLOCK, 0, stream>>>(
      accum_ws, w16s, b_out, (float*)d_out, N);
}
